// Arch19GraphEncoder_25288767438902
// MI455X (gfx1250) — compile-verified, matrix operand order checked by reading, not run on hardware
//
#include <hip/hip_runtime.h>
#include <hip/hip_bf16.h>

// ---------------------------------------------------------------------------
// Arch19 graph encoder forward for MI455X (gfx1250, wave32, WMMA).
// All dense GEMMs (input proj, Q/K/V, Wo, FF1, FF2 for intra+inter layers)
// go through v_wmma_f32_16x16x32_f16 (f16 inputs, f32 accumulate).
// ---------------------------------------------------------------------------

#define BGRAPH 2048
#define NPER   32
#define NNODE  (BGRAPH * NPER)      // 65536
#define EEDGE  196608
#define MSUB   32
#define SSUB   (BGRAPH * MSUB)      // 65536
#define KSUB   10
#define HD     256
#define NHEAD  4
#define DHEAD  64
#define RWSTEP 16
#define ESAMP  393216
#define SK     (SSUB * KSUB)        // 655360
#define FFD    1024

typedef __attribute__((ext_vector_type(16))) _Float16 v16h;
typedef __attribute__((ext_vector_type(8)))  float    v8f;

union FragH { v16h v; unsigned int u[8]; };
union PackH { _Float16 h[2]; unsigned int u; };

// ---------------------------------------------------------------------------
// Generic WMMA GEMM: Out[M,N] = act(A[M,K](f16) @ Bw[K,N](f32) + bias [+ Res])
// Block = 256 threads (8 waves), 128x64 tile, K-chunk 32.
// Each wave owns a 32x32 quadrant = 2x2 WMMA tiles.
// ---------------------------------------------------------------------------
template <bool RELU, bool RESID, typename OutT>
__global__ __launch_bounds__(256)
void gemm_wmma(const _Float16* __restrict__ A, const float* __restrict__ Bw,
               const float* __restrict__ bias, const float* __restrict__ Res,
               OutT* __restrict__ Out, int M, int N, int Kd)
{
    __shared__ _Float16 As[128][40];   // [row][k]
    __shared__ _Float16 BsT[64][40];   // [col][k]  (transposed stage)

    const int bm = blockIdx.x * 128;
    const int bn = blockIdx.y * 64;
    const int tid  = threadIdx.x;
    const int lane = tid & 31;
    const int wave = tid >> 5;
    const int mq = (wave >> 1) * 32;   // 0,32,64,96
    const int nq = (wave & 1) * 32;    // 0,32
    const int lm  = lane & 15;
    const int khi = (lane >> 4) << 3;  // 0 or 8 (A/B fragment k-offset)
    const int rlo = (lane >> 4) << 3;  // C-tile row offset

    v8f acc[2][2];
    #pragma unroll
    for (int ti = 0; ti < 2; ++ti)
        #pragma unroll
        for (int tj = 0; tj < 2; ++tj)
            acc[ti][tj] = v8f{0.f, 0.f, 0.f, 0.f, 0.f, 0.f, 0.f, 0.f};

    const int ar = tid >> 1;           // A-stage row (0..127)
    const int ahb = (tid & 1) * 16;    // A-stage half-base

    for (int k0 = 0; k0 < Kd; k0 += 32) {
        // stage A tile: 128 rows x 32 halves (32B per thread)
        {
            const unsigned int* src =
                (const unsigned int*)(A + (size_t)(bm + ar) * Kd + k0 + ahb);
            unsigned int* dst = (unsigned int*)(&As[ar][ahb]);
            #pragma unroll
            for (int u = 0; u < 8; ++u) dst[u] = src[u];
        }
        // stage B tile transposed + packed: BsT[col][kk..kk+1], one b32 store
        #pragma unroll
        for (int e = 0; e < 4; ++e) {
            const int idx = tid + e * 256;      // 0..1023 (pairs)
            const int col = idx & 63;
            const int kk  = (idx >> 6) << 1;    // 0,2,..,30
            PackH p;
            p.h[0] = (_Float16)Bw[(size_t)(k0 + kk) * N + bn + col];
            p.h[1] = (_Float16)Bw[(size_t)(k0 + kk + 1) * N + bn + col];
            *(unsigned int*)&BsT[col][kk] = p.u;
        }
        // prefetch next A K-chunk while we compute this one
        if (k0 + 32 < Kd)
            __builtin_prefetch(A + (size_t)(bm + ar) * Kd + k0 + 32 + ahb, 0, 0);
        __syncthreads();

        FragH af[2], bf[2];
        #pragma unroll
        for (int ti = 0; ti < 2; ++ti) {
            const int row = mq + ti * 16 + lm;
            #pragma unroll
            for (int j = 0; j < 8; ++j) {
                const int kb = ((j >> 2) << 4) + ((j & 3) << 1) + khi;
                af[ti].u[j] = *(const unsigned int*)&As[row][kb];
            }
        }
        #pragma unroll
        for (int tj = 0; tj < 2; ++tj) {
            const int col = nq + tj * 16 + lm;
            #pragma unroll
            for (int j = 0; j < 8; ++j) {
                const int kb = ((j >> 2) << 4) + ((j & 3) << 1) + khi;
                bf[tj].u[j] = *(const unsigned int*)&BsT[col][kb];
            }
        }
        #pragma unroll
        for (int ti = 0; ti < 2; ++ti)
            #pragma unroll
            for (int tj = 0; tj < 2; ++tj)
                acc[ti][tj] = __builtin_amdgcn_wmma_f32_16x16x32_f16(
                    false, af[ti].v, false, bf[tj].v,
                    (short)0, acc[ti][tj], false, false);
        __syncthreads();
    }

    // epilogue: C layout -> row = base + vgpr + 8*(lane>=16), col = base + lane%16
    #pragma unroll
    for (int ti = 0; ti < 2; ++ti) {
        const int growb = bm + mq + ti * 16 + rlo;
        #pragma unroll
        for (int tj = 0; tj < 2; ++tj) {
            const int gcol = bn + nq + tj * 16 + lm;
            const float bv = bias[gcol];
            OutT* op = Out + (size_t)growb * N + gcol;
            const float* rp = nullptr;
            if constexpr (RESID) rp = Res + (size_t)growb * N + gcol;
            #pragma unroll
            for (int r = 0; r < 8; ++r) {
                float v = acc[ti][tj][r] + bv;
                if constexpr (RESID) v += rp[(size_t)r * N];
                if constexpr (RELU)  v = fmaxf(v, 0.f);
                op[(size_t)r * N] = (OutT)v;
            }
        }
    }
}

// ---------------------------------------------------------------------------
// RWSE: dense adjacency build + per-graph 32x32 power iteration
// ---------------------------------------------------------------------------
__global__ void build_adense(const int* __restrict__ ei, float* __restrict__ Ad)
{
    const int e = blockIdx.x * blockDim.x + threadIdx.x;
    if (e >= EEDGE) return;
    const int s = ei[e];
    const int d = ei[EEDGE + e];
    const int g = s >> 5;
    Ad[(size_t)g * 1024 + (size_t)(s & 31) * 32 + (d & 31)] = 1.f;
}

__global__ __launch_bounds__(256)
void rwse_power(const float* __restrict__ Ad, float* __restrict__ rw)
{
    __shared__ float T[32][33];
    __shared__ float Pm[32][33];
    __shared__ float Pn[32][33];
    __shared__ float deg[32];
    const int g = blockIdx.x;
    const int t = threadIdx.x;
    const float* Ag = Ad + (size_t)g * 1024;
    if (t < 32) {
        float s = 0.f;
        for (int j = 0; j < 32; ++j) s += Ag[t * 32 + j];
        deg[t] = fmaxf(s, 1.f);
    }
    __syncthreads();
    #pragma unroll
    for (int e = 0; e < 4; ++e) {
        const int idx = t + e * 256;
        const int i = idx >> 5, j = idx & 31;
        T[i][j]  = Ag[i * 32 + j] / deg[i];
        Pm[i][j] = (i == j) ? 1.f : 0.f;
    }
    __syncthreads();
    for (int step = 0; step < RWSTEP; ++step) {
        #pragma unroll
        for (int e = 0; e < 4; ++e) {
            const int idx = t + e * 256;
            const int i = idx >> 5, j = idx & 31;
            float a = 0.f;
            for (int kk = 0; kk < 32; ++kk) a += Pm[i][kk] * T[kk][j];
            Pn[i][j] = a;
        }
        __syncthreads();
        #pragma unroll
        for (int e = 0; e < 4; ++e) {
            const int idx = t + e * 256;
            const int i = idx >> 5, j = idx & 31;
            Pm[i][j] = Pn[i][j];
            if (i == j) rw[((size_t)g * 32 + i) * RWSTEP + step] = Pn[i][i];
        }
        __syncthreads();
    }
}

__global__ void rwse_embed(const float* __restrict__ rw, const float* __restrict__ W,
                           const float* __restrict__ b, float* __restrict__ out)
{
    __shared__ float r[RWSTEP];
    const int node = blockIdx.x;
    const int c = threadIdx.x;
    if (c < RWSTEP) r[c] = rw[(size_t)node * RWSTEP + c];
    __syncthreads();
    float acc = b[c];
    #pragma unroll
    for (int t = 0; t < RWSTEP; ++t) acc += r[t] * W[t * HD + c];
    out[(size_t)node * HD + c] = fmaxf(acc, 0.f);
}

// ---------------------------------------------------------------------------
// Sampled-edge processing
// ---------------------------------------------------------------------------
__global__ __launch_bounds__(256)
void bond_scatter(const int* __restrict__ eis, const int* __restrict__ esg,
                  const int* __restrict__ eattr, const float* __restrict__ bond,
                  float* __restrict__ sums, float* __restrict__ cnt)
{
    const int w = (blockIdx.x * 256 + threadIdx.x) >> 5;  // wave per edge
    const int lane = threadIdx.x & 31;
    if (w >= ESAMP) return;
    const int d  = eis[ESAMP + w];
    const int ge = esg[w];
    const int bt = eattr[ge] - 1;
    const float* bf = bond + (size_t)bt * HD;
    float* sp = sums + (size_t)d * HD;
    #pragma unroll
    for (int i = 0; i < 8; ++i) atomicAdd(&sp[lane + 32 * i], bf[lane + 32 * i]);
    if (lane == 0) atomicAdd(&cnt[d], 1.f);
}

__global__ void build_asub(const int* __restrict__ eis, float* __restrict__ Asub)
{
    const int e = blockIdx.x * blockDim.x + threadIdx.x;
    if (e >= ESAMP) return;
    const int s = eis[e];
    const int d = eis[ESAMP + e];
    const int sub = s / KSUB;
    Asub[(size_t)sub * (KSUB * KSUB) + (size_t)(s % KSUB) * KSUB + (d % KSUB)] = 1.f;
}

__global__ void build_xcat(const int* __restrict__ nodes, const int* __restrict__ x,
                           const float* __restrict__ atomT, const float* __restrict__ rwemb,
                           const float* __restrict__ sums, const float* __restrict__ cnt,
                           _Float16* __restrict__ Xc)
{
    const int row = blockIdx.x;
    const int c   = threadIdx.x;
    const int nd  = nodes[row];
    const int node = nd < 0 ? 0 : nd;
    const int xi  = x[node];
    const size_t o = (size_t)row * (3 * HD);
    Xc[o + c]          = (_Float16)atomT[(size_t)xi * HD + c];
    Xc[o + HD + c]     = (_Float16)rwemb[(size_t)node * HD + c];
    float cv = cnt[row];
    cv = cv < 1.f ? 1.f : cv;
    Xc[o + 2 * HD + c] = (_Float16)(sums[(size_t)row * HD + c] / cv);
}

__global__ void add_pe_mask(float* __restrict__ h, const float* __restrict__ Asub,
                            const float* __restrict__ adjW, const float* __restrict__ pos,
                            const int* __restrict__ nodes)
{
    const int row = blockIdx.x;
    const int c   = threadIdx.x;
    const int s = row / KSUB, ks = row % KSUB;
    float acc = h[(size_t)row * HD + c] + pos[ks * HD + c];
    const float* ar = Asub + (size_t)s * (KSUB * KSUB) + ks * KSUB;
    #pragma unroll
    for (int j = 0; j < KSUB; ++j) acc += ar[j] * adjW[j * HD + c];
    const float vf = (nodes[row] >= 0) ? 1.f : 0.f;
    h[(size_t)row * HD + c] = acc * vf;
}

// ---------------------------------------------------------------------------
// LayerNorm over H=256, wave-per-row
// ---------------------------------------------------------------------------
template <typename OutT>
__global__ __launch_bounds__(256)
void ln_rows(const float* __restrict__ X, const float* __restrict__ g,
             const float* __restrict__ b, OutT* __restrict__ Y, int Mrows)
{
    const int wave = threadIdx.x >> 5;
    const int lane = threadIdx.x & 31;
    const int row = blockIdx.x * 8 + wave;
    if (row >= Mrows) return;
    const float* xr = X + (size_t)row * HD;
    float vals[8];
    float s = 0.f;
    #pragma unroll
    for (int i = 0; i < 8; ++i) { vals[i] = xr[lane + 32 * i]; s += vals[i]; }
    #pragma unroll
    for (int m = 16; m; m >>= 1) s += __shfl_xor(s, m, 32);
    const float mean = s * (1.f / HD);
    float vs = 0.f;
    #pragma unroll
    for (int i = 0; i < 8; ++i) { const float d = vals[i] - mean; vs += d * d; }
    #pragma unroll
    for (int m = 16; m; m >>= 1) vs += __shfl_xor(vs, m, 32);
    const float inv = rsqrtf(vs * (1.f / HD) + 1e-5f);
    #pragma unroll
    for (int i = 0; i < 8; ++i) {
        const int c = lane + 32 * i;
        Y[(size_t)row * HD + c] = (OutT)((vals[i] - mean) * inv * g[c] + b[c]);
    }
}

// ---------------------------------------------------------------------------
// Small-sequence MHA (L = 10 or 32, NH=4, DH=64), one block per group
// ---------------------------------------------------------------------------
__global__ __launch_bounds__(128)
void attn_small(const float* __restrict__ Q, const float* __restrict__ Kt,
                const float* __restrict__ Vt, _Float16* __restrict__ O, int L)
{
    __shared__ _Float16 Ks[32][260];
    __shared__ _Float16 Vs[32][260];
    __shared__ float    Lg[NHEAD][32][32];
    const size_t base = (size_t)blockIdx.x * L;
    const int t = threadIdx.x;
    for (int idx = t; idx < L * HD; idx += 128) {
        const int r = idx >> 8, c = idx & 255;
        Ks[r][c] = (_Float16)Kt[(base + r) * HD + c];
        Vs[r][c] = (_Float16)Vt[(base + r) * HD + c];
    }
    __syncthreads();
    const int LL = L * L;
    for (int idx = t; idx < NHEAD * LL; idx += 128) {
        const int hh = idx / LL, rem = idx % LL;
        const int qi = rem / L, ki = rem % L;
        const float* qp = Q + (base + qi) * HD + hh * DHEAD;
        float acc = 0.f;
        #pragma unroll 8
        for (int d = 0; d < DHEAD; ++d) acc += qp[d] * (float)Ks[ki][hh * DHEAD + d];
        Lg[hh][qi][ki] = acc * 0.125f;  // 1/sqrt(64)
    }
    __syncthreads();
    if (t < NHEAD * L) {
        const int hh = t / L, qi = t % L;
        float mx = -1e30f;
        for (int ki = 0; ki < L; ++ki) mx = fmaxf(mx, Lg[hh][qi][ki]);
        float sm = 0.f;
        for (int ki = 0; ki < L; ++ki) {
            const float e = __expf(Lg[hh][qi][ki] - mx);
            Lg[hh][qi][ki] = e; sm += e;
        }
        const float inv = 1.f / sm;
        for (int ki = 0; ki < L; ++ki) Lg[hh][qi][ki] *= inv;
    }
    __syncthreads();
    for (int idx = t; idx < L * HD; idx += 128) {
        const int qi = idx >> 8, c = idx & 255, hh = c >> 6;
        float acc = 0.f;
        for (int ki = 0; ki < L; ++ki) acc += Lg[hh][qi][ki] * (float)Vs[ki][c];
        O[(base + qi) * HD + c] = (_Float16)acc;
    }
}

// ---------------------------------------------------------------------------
// Poolings
// ---------------------------------------------------------------------------
__global__ void meanpool_masked(const float* __restrict__ lnT, const int* __restrict__ nodes,
                                float* __restrict__ hsub)
{
    const int s = blockIdx.x, c = threadIdx.x;
    float acc = 0.f, cv = 0.f;
    for (int kk = 0; kk < KSUB; ++kk) {
        const float vf = (nodes[s * KSUB + kk] >= 0) ? 1.f : 0.f;
        acc += lnT[((size_t)s * KSUB + kk) * HD + c] * vf;
        cv  += vf;
    }
    cv = cv < 1.f ? 1.f : cv;
    hsub[(size_t)s * HD + c] = acc / cv;
}

__global__ void meanpool_plain(const float* __restrict__ lnT, float* __restrict__ out)
{
    const int b = blockIdx.x, c = threadIdx.x;
    float acc = 0.f;
    for (int r = 0; r < MSUB; ++r) acc += lnT[((size_t)b * MSUB + r) * HD + c];
    out[(size_t)b * HD + c] = acc * (1.f / MSUB);
}

// ---------------------------------------------------------------------------
// Host side
// ---------------------------------------------------------------------------
struct LayerP {
    const float *W1, *W2, *Wk, *Wo, *Wq, *Wv;
    const float *b1, *b2, *bk, *bo, *bq, *bv;
    const float *ln1b, *ln1g, *ln2b, *ln2g;
};

static LayerP get_layer(void* const* d_in, int base)
{
    LayerP L;
    L.W1   = (const float*)d_in[base + 0];
    L.W2   = (const float*)d_in[base + 1];
    L.Wk   = (const float*)d_in[base + 2];
    L.Wo   = (const float*)d_in[base + 3];
    L.Wq   = (const float*)d_in[base + 4];
    L.Wv   = (const float*)d_in[base + 5];
    L.b1   = (const float*)d_in[base + 6];
    L.b2   = (const float*)d_in[base + 7];
    L.bk   = (const float*)d_in[base + 8];
    L.bo   = (const float*)d_in[base + 9];
    L.bq   = (const float*)d_in[base + 10];
    L.bv   = (const float*)d_in[base + 11];
    L.ln1b = (const float*)d_in[base + 12];
    L.ln1g = (const float*)d_in[base + 13];
    L.ln2b = (const float*)d_in[base + 14];
    L.ln2g = (const float*)d_in[base + 15];
    return L;
}

extern "C" void kernel_launch(void* const* d_in, const int* in_sizes, int n_in,
                              void* d_out, int out_size, void* d_ws, size_t ws_size,
                              hipStream_t stream)
{
    (void)in_sizes; (void)n_in; (void)out_size; (void)ws_size;

    // ---- inputs (setup_inputs() dict order) ----
    const int* x          = (const int*)d_in[0];
    const int* edge_attr  = (const int*)d_in[1];
    const int* edge_index = (const int*)d_in[2];
    const int* nodes      = (const int*)d_in[5];
    const int* eis        = (const int*)d_in[6];
    const int* esg        = (const int*)d_in[7];
    // params: jax pytree flatten = sorted dict keys, tuples in order.
    // adj_W, atom, bond, inp_W, inp_b, inter(2x16), inter_nb, inter_ng,
    // intra(2x16), intra_nb, intra_ng, pos, rwse_W, rwse_b
    const float* adjW  = (const float*)d_in[8];
    const float* atomT = (const float*)d_in[9];
    const float* bondT = (const float*)d_in[10];
    const float* inpW  = (const float*)d_in[11];
    const float* inpB  = (const float*)d_in[12];
    LayerP inter0 = get_layer(d_in, 13);
    LayerP inter1 = get_layer(d_in, 29);
    const float* inter_nb = (const float*)d_in[45];
    const float* inter_ng = (const float*)d_in[46];
    LayerP intra0 = get_layer(d_in, 47);
    LayerP intra1 = get_layer(d_in, 63);
    const float* intra_nb = (const float*)d_in[79];
    const float* intra_ng = (const float*)d_in[80];
    const float* posT  = (const float*)d_in[81];
    const float* rwseW = (const float*)d_in[82];
    const float* rwseB = (const float*)d_in[83];

    // ---- workspace layout ----
    char* w = (char*)d_ws;
    size_t off = 0;
    auto alloc = [&](size_t bytes) {
        size_t o = off;
        off += (bytes + 255) & ~(size_t)255;
        return o;
    };
    const size_t o_adense = alloc((size_t)BGRAPH * 1024 * 4);
    const size_t o_rwse   = alloc((size_t)NNODE * RWSTEP * 4);
    const size_t o_rwemb  = alloc((size_t)NNODE * HD * 4);
    const size_t o_asub   = alloc((size_t)SSUB * KSUB * KSUB * 4);
    const size_t o_cnt    = alloc((size_t)SK * 4);
    const size_t o_q      = alloc((size_t)SK * HD * 4);   // aliases bond sums
    const size_t o_k      = alloc((size_t)SK * HD * 4);
    const size_t o_v      = alloc((size_t)SK * HD * 4);
    const size_t o_h      = alloc((size_t)SK * HD * 4);   // residual stream (intra)
    const size_t o_Y      = alloc((size_t)SK * HD * 2);   // f16 GEMM input
    const size_t o_attn   = alloc((size_t)SK * HD * 2);   // f16 attention out
    const size_t o_F      = alloc((size_t)SK * FFD * 2);  // f16 FF hidden
    const size_t o_xcat   = alloc((size_t)SK * 3 * HD * 2); // aliases final-LN temp
    const size_t o_hsub   = alloc((size_t)SSUB * HD * 4); // residual stream (inter)

    float*     Ad    = (float*)(w + o_adense);
    float*     Rw    = (float*)(w + o_rwse);
    float*     Rwemb = (float*)(w + o_rwemb);
    float*     Asub  = (float*)(w + o_asub);
    float*     Cnt   = (float*)(w + o_cnt);
    float*     Qb    = (float*)(w + o_q);
    float*     Sums  = (float*)(w + o_q);    // alias: used before Q is needed
    float*     Kb    = (float*)(w + o_k);
    float*     Vb    = (float*)(w + o_v);
    float*     Hb    = (float*)(w + o_h);
    _Float16*  Yh    = (_Float16*)(w + o_Y);
    _Float16*  Ah    = (_Float16*)(w + o_attn);
    _Float16*  Fh    = (_Float16*)(w + o_F);
    _Float16*  Xc    = (_Float16*)(w + o_xcat);
    float*     LnT   = (float*)(w + o_xcat); // alias: Xcat consumed before final LN
    float*     Hsub  = (float*)(w + o_hsub);
    float*     Outf  = (float*)d_out;

    // ---- zero-init scatter targets ----
    hipMemsetAsync(Ad,   0, (size_t)BGRAPH * 1024 * 4, stream);
    hipMemsetAsync(Sums, 0, (size_t)SK * HD * 4, stream);
    hipMemsetAsync(Cnt,  0, (size_t)SK * 4, stream);
    hipMemsetAsync(Asub, 0, (size_t)SSUB * KSUB * KSUB * 4, stream);

    // ---- RWSE ----
    build_adense<<<(EEDGE + 255) / 256, 256, 0, stream>>>(edge_index, Ad);
    rwse_power<<<BGRAPH, 256, 0, stream>>>(Ad, Rw);
    rwse_embed<<<NNODE, HD, 0, stream>>>(Rw, rwseW, rwseB, Rwemb);

    // ---- sampled-edge features ----
    bond_scatter<<<ESAMP / 8, 256, 0, stream>>>(eis, esg, edge_attr, bondT, Sums, Cnt);
    build_asub<<<(ESAMP + 255) / 256, 256, 0, stream>>>(eis, Asub);

    // ---- input projection: concat(768) @ inp_W -> h ----
    build_xcat<<<SK, HD, 0, stream>>>(nodes, x, atomT, Rwemb, Sums, Cnt, Xc);
    {
        dim3 g(SK / 128, HD / 64);
        gemm_wmma<false, false, float><<<g, 256, 0, stream>>>(
            Xc, inpW, inpB, nullptr, Hb, SK, HD, 3 * HD);
    }
    add_pe_mask<<<SK, HD, 0, stream>>>(Hb, Asub, adjW, posT, nodes);

    // ---- transformer layer runner ----
    auto run_layer = [&](const LayerP& L, float* res, int Mrows, int Lseq, int groups) {
        dim3 gH(Mrows / 128, HD / 64);
        dim3 gF(Mrows / 128, FFD / 64);
        const int lnB = (Mrows + 7) / 8;
        ln_rows<_Float16><<<lnB, 256, 0, stream>>>(res, L.ln1g, L.ln1b, Yh, Mrows);
        gemm_wmma<false, false, float><<<gH, 256, 0, stream>>>(Yh, L.Wq, L.bq, nullptr, Qb, Mrows, HD, HD);
        gemm_wmma<false, false, float><<<gH, 256, 0, stream>>>(Yh, L.Wk, L.bk, nullptr, Kb, Mrows, HD, HD);
        gemm_wmma<false, false, float><<<gH, 256, 0, stream>>>(Yh, L.Wv, L.bv, nullptr, Vb, Mrows, HD, HD);
        attn_small<<<groups, 128, 0, stream>>>(Qb, Kb, Vb, Ah, Lseq);
        gemm_wmma<false, true, float><<<gH, 256, 0, stream>>>(Ah, L.Wo, L.bo, res, res, Mrows, HD, HD);
        ln_rows<_Float16><<<lnB, 256, 0, stream>>>(res, L.ln2g, L.ln2b, Yh, Mrows);
        gemm_wmma<true, false, _Float16><<<gF, 256, 0, stream>>>(Yh, L.W1, L.b1, nullptr, Fh, Mrows, FFD, HD);
        gemm_wmma<false, true, float><<<gH, 256, 0, stream>>>(Fh, L.W2, L.b2, res, res, Mrows, HD, FFD);
    };

    // ---- intra layers (tokens = S*K, groups of K=10) ----
    run_layer(intra0, Hb, SK, KSUB, SSUB);
    run_layer(intra1, Hb, SK, KSUB, SSUB);

    // ---- intra final LN + masked mean pool over K ----
    ln_rows<float><<<(SK + 7) / 8, 256, 0, stream>>>(Hb, intra_ng, intra_nb, LnT, SK);
    meanpool_masked<<<SSUB, HD, 0, stream>>>(LnT, nodes, Hsub);

    // ---- inter layers (tokens = S, groups of M=32) ----
    run_layer(inter0, Hsub, SSUB, MSUB, SSUB / MSUB);
    run_layer(inter1, Hsub, SSUB, MSUB, SSUB / MSUB);

    // ---- inter final LN + mean pool over M -> [B, H] ----
    ln_rows<float><<<(SSUB + 7) / 8, 256, 0, stream>>>(Hsub, inter_ng, inter_nb, LnT, SSUB);
    meanpool_plain<<<BGRAPH, HD, 0, stream>>>(LnT, Outf);
}
